// ScaledDotProdSelfAtt_43293270343833
// MI455X (gfx1250) — matrix operand order repaired and checked by reading, not compiled, not run on hardware
//
#include <hip/hip_runtime.h>
#include <cstdint>

#define B_ 8
#define S_ 2048
#define E_ 1024
#define H_ 64

typedef __attribute__((ext_vector_type(16))) _Float16 v16h;
typedef __attribute__((ext_vector_type(4)))  _Float16 v4h;
typedef __attribute__((ext_vector_type(8)))  float    v8f;
typedef __attribute__((ext_vector_type(4)))  unsigned int v4u;
typedef __attribute__((ext_vector_type(8)))  int      v8i;
typedef __attribute__((ext_vector_type(4)))  int      v4i;

#if defined(__has_builtin)
#  if __has_builtin(__builtin_amdgcn_tensor_load_to_lds) && __has_builtin(__builtin_amdgcn_s_wait_tensorcnt)
#    define TDM_OK 1
#  endif
#endif
#ifndef TDM_OK
#  define TDM_OK 0
#endif

static __device__ __forceinline__ v8f wmma_f32_f16(v16h a, v16h b, v8f c) {
  return __builtin_amdgcn_wmma_f32_16x16x32_f16(false, a, false, b, (short)0, c, false, false);
}

#if TDM_OK
// 1-D contiguous TDM load: nelem f16 elements global -> LDS.
// Descriptor fields per CDNA5 ISA (all byte-aligned).
static __device__ __forceinline__ void tdm_load_1d(uint32_t lds_off, uint64_t gaddr,
                                                   uint32_t nelem) {
  v4u g0;
  g0[0] = 1u;                                   // count=1, user mode
  g0[1] = lds_off;                              // lds_addr
  g0[2] = (uint32_t)gaddr;                      // global_addr[31:0]
  g0[3] = (uint32_t)((gaddr >> 32) & 0x01FFFFFFu) | (2u << 30);  // addr[56:32] | type=2
  v8i g1;
  g1[0] = 0x10000;                              // data_size=2B @ [17:16]
  g1[1] = (int)(nelem << 16);                   // tensor_dim0[15:0] @ [63:48]
  g1[2] = (int)((nelem >> 16) | (1u << 16));    // tensor_dim0[31:16] | tensor_dim1=1
  g1[3] = (int)(nelem << 16);                   // tile_dim0 @ [127:112]
  g1[4] = 1;                                    // tile_dim1=1
  g1[5] = (int)nelem;                           // tensor_dim0_stride[31:0]
  g1[6] = 0;
  g1[7] = 0;
  v4i z4 = {0, 0, 0, 0};
#if __clang_major__ >= 23
  v8i z8 = {0, 0, 0, 0, 0, 0, 0, 0};
  __builtin_amdgcn_tensor_load_to_lds(g0, g1, z4, z4, z8, 0);
#else
  __builtin_amdgcn_tensor_load_to_lds(g0, g1, z4, z4, 0);
#endif
}
#endif

// ---------------- weight swizzle: f32 [E,H] -> f16 fragment-order ----------------
// Layout: step ks (32 K rows) -> 384 blocks of 16 halfs:
//   block = mat*128 + nt*32 + lane,  half h:  element W[ks*32 + (lane>>4)*16 + h][nt*16 + (lane&15)]
__global__ void swz_w_kernel(const float* __restrict__ wq, const float* __restrict__ wk,
                             const float* __restrict__ wv, _Float16* __restrict__ wsw) {
  int idx = blockIdx.x * 256 + threadIdx.x;     // 0 .. 196607
  int h = idx & 15;
  int blk16 = idx >> 4;
  int ks = blk16 / 384;
  int b2 = blk16 - ks * 384;
  int mat = b2 >> 7;
  int r2 = b2 & 127;
  int nt = r2 >> 5;
  int lane = r2 & 31;
  int k = ks * 32 + (lane >> 4) * 16 + h;
  int n = nt * 16 + (lane & 15);
  const float* src = (mat == 0) ? wq : (mat == 1) ? wk : wv;
  wsw[idx] = (_Float16)src[(size_t)k * H_ + n];
}

// ---------------- fused QKV projection ----------------
// grid: (B*S)/128 blocks, 256 threads = 8 waves. Outputs Q/K/V pre-swizzled
// into WMMA fragment order for the attention kernel.
__global__ __launch_bounds__(256) void qkv_kernel(
    const float* __restrict__ x, const _Float16* __restrict__ wsw,
    _Float16* __restrict__ qsw, _Float16* __restrict__ kvsw) {
  __shared__ alignas(32) _Float16 lds_xa[4096];       // x tile, A-fragment order
  __shared__ alignas(32) _Float16 lds_wf[2][6144];    // weight tiles, B-fragment order

  const int t    = threadIdx.x;
  const int wid  = t >> 5;
  const int lane = t & 31;
  const int ln   = lane & 15;
  const int hi   = (lane >> 4) & 1;
  const int row0 = blockIdx.x * 128;

  v8f acc[3][4];
  #pragma unroll
  for (int m = 0; m < 3; ++m)
    #pragma unroll
    for (int n = 0; n < 4; ++n) acc[m][n] = {};

#if TDM_OK
  const uint32_t lds_w_off = (uint32_t)(uintptr_t)&lds_wf[0][0];
  if (wid == 0)
    tdm_load_1d(lds_w_off, (uint64_t)(uintptr_t)wsw, 6144);
#endif

  #pragma unroll 2
  for (int ks = 0; ks < E_ / 32; ++ks) {
    const int kb = ks * 32;
#if TDM_OK
    if (wid == 0) {
      if (ks + 1 < E_ / 32) {
        tdm_load_1d(lds_w_off + ((ks + 1) & 1) * 12288u,
                    (uint64_t)(uintptr_t)(wsw + (size_t)(ks + 1) * 6144), 6144);
        __builtin_amdgcn_s_wait_tensorcnt(1);
      } else {
        __builtin_amdgcn_s_wait_tensorcnt(0);
      }
    }
#else
    #pragma unroll
    for (int i = 0; i < 3; ++i) {
      int id = t + 256 * i;   // 768 uint4 = 12KB
      ((uint4*)lds_wf[ks & 1])[id] = ((const uint4*)(wsw + (size_t)ks * 6144))[id];
    }
#endif
    // stage x tile [128 x 32] f32 -> f16 LDS in A-fragment order
    #pragma unroll
    for (int i = 0; i < 4; ++i) {
      int id = t + 256 * i;           // 0..1023 float4 slots
      int r = id >> 3, g = id & 7;
      int k0 = g * 4;
      const float4 f = *(const float4*)(x + (size_t)(row0 + r) * E_ + kb + k0);
      int wv_r = r >> 4, ln_r = r & 15;
      int hi_w = k0 >> 4;
      int h0 = k0 & 15;
      v4h d;
      d[0] = (_Float16)f.x; d[1] = (_Float16)f.y;
      d[2] = (_Float16)f.z; d[3] = (_Float16)f.w;
      *(v4h*)&lds_xa[((wv_r * 32 + hi_w * 16 + ln_r) << 4) + h0] = d;
    }
    if (ks + 1 < E_ / 32)
      __builtin_prefetch(x + (size_t)(row0 + (t >> 3)) * E_ + kb + 32 + (t & 7) * 4, 0, 3);
    __syncthreads();

    const v16h a = *(const v16h*)&lds_xa[(wid * 32 + lane) << 4];
    const _Float16* wf = lds_wf[ks & 1];
    #pragma unroll
    for (int mat = 0; mat < 3; ++mat) {
      #pragma unroll
      for (int nt = 0; nt < 4; ++nt) {
        const v16h b = *(const v16h*)&wf[((mat * 4 + nt) * 32 + lane) << 4];
        acc[mat][nt] = wmma_f32_f16(a, b, acc[mat][nt]);
      }
    }
    __syncthreads();
  }

  // epilogue: scatter into pre-swizzled Q (A-frag) and K/V (B-frag) layouts
  #pragma unroll
  for (int nt = 0; nt < 4; ++nt) {
    #pragma unroll
    for (int r = 0; r < 8; ++r) {
      const int grow = row0 + wid * 16 + r + hi * 8;   // global row in [0, B*S)
      const int col  = nt * 16 + ln;                   // head dim
      const int b    = grow >> 11;                     // batch
      const int s    = grow & 2047;                    // position in batch
      // Q: A-fragment layout per 16-row block
      {
        int rb = grow >> 4, rl = grow & 15;
        int f = col >> 5, c = col & 31;
        int hia = (c >> 3) & 1, ha = (c & 7) + ((c & 16) >> 1);
        qsw[(size_t)rb * 1024 + ((f * 32 + hia * 16 + rl) << 4) + ha] =
            (_Float16)acc[0][nt][r];
      }
      // K: B-fragment layout for Q*K^T, interleaved tile [K(2048) | V(2048)]
      {
        int kt = s >> 5;
        int ntk = (s >> 4) & 1, lnk = s & 15;
        int f = col >> 5, c = col & 31;
        int hik = (c >> 3) & 1, hk = (c & 7) + ((c & 16) >> 1);
        kvsw[(size_t)b * 262144 + (size_t)kt * 4096 +
             (((f * 2 + ntk) * 32 + hik * 16 + lnk) << 4) + hk] =
            (_Float16)acc[1][nt][r];
      }
      // V: B-fragment layout for P*V
      {
        int kt = s >> 5;
        int kk = s & 31;
        int hiv = (kk >> 3) & 1, hv = (kk & 7) + ((kk & 16) >> 1);
        int ntv = col >> 4, lnv = col & 15;
        kvsw[(size_t)b * 262144 + (size_t)kt * 4096 + 2048 +
             ((ntv * 32 + hiv * 16 + lnv) << 4) + hv] =
            (_Float16)acc[2][nt][r];
      }
    }
  }
}

// ---------------- causal flash attention ----------------
// grid: (S/128, B), 256 threads = 8 waves, each wave owns 16 query rows.
__global__ __launch_bounds__(256) void attn_kernel(
    const _Float16* __restrict__ qsw, const _Float16* __restrict__ kvsw,
    float* __restrict__ out) {
  __shared__ alignas(32) _Float16 lds_kv[2][4096];    // [K frags | V frags] per tile
  __shared__ alignas(32) _Float16 lds_pw[8][512];     // per-wave P in A-frag order

  const int t    = threadIdx.x;
  const int wid  = t >> 5;
  const int lane = t & 31;
  const int ln   = lane & 15;
  const int hi   = (lane >> 4) & 1;
  const int q0   = blockIdx.x * 128;
  const int bb   = blockIdx.y * S_;
  const int Qr   = q0 + wid * 16;
  const float scale = 0.03125f;                       // 1024^-0.5

  // Q fragments: contiguous A-fragment blocks
  const int rbq = (bb + Qr) >> 4;
  const v16h aq0 = *(const v16h*)(qsw + (size_t)rbq * 1024 + ((0 * 32 + lane) << 4));
  const v16h aq1 = *(const v16h*)(qsw + (size_t)rbq * 1024 + ((1 * 32 + lane) << 4));

  float mrow[8], lrow[8];
  v8f accO[4];
  #pragma unroll
  for (int r = 0; r < 8; ++r) { mrow[r] = -1e30f; lrow[r] = 0.f; }
  #pragma unroll
  for (int n = 0; n < 4; ++n) accO[n] = {};

  const _Float16* kvbase = kvsw + (size_t)blockIdx.y * 262144;
  const int nkt = blockIdx.x * 4 + 4;

#if TDM_OK
  const uint32_t lds_kv_off = (uint32_t)(uintptr_t)&lds_kv[0][0];
  if (wid == 0)
    tdm_load_1d(lds_kv_off, (uint64_t)(uintptr_t)kvbase, 4096);
#endif

  for (int kt = 0; kt < nkt; ++kt) {
    const int kb = kt * 32;
#if TDM_OK
    if (wid == 0) {
      if (kt + 1 < nkt) {
        tdm_load_1d(lds_kv_off + ((kt + 1) & 1) * 8192u,
                    (uint64_t)(uintptr_t)(kvbase + (size_t)(kt + 1) * 4096), 4096);
        __builtin_amdgcn_s_wait_tensorcnt(1);
      } else {
        __builtin_amdgcn_s_wait_tensorcnt(0);
      }
    }
#else
    #pragma unroll
    for (int i = 0; i < 2; ++i) {
      int id = t + 256 * i;   // 512 uint4 = 8KB
      ((uint4*)lds_kv[kt & 1])[id] = ((const uint4*)(kvbase + (size_t)kt * 4096))[id];
    }
    if (kt + 1 < nkt)
      __builtin_prefetch(kvbase + (size_t)(kt + 1) * 4096 + t * 16, 0, 3);
#endif
    __syncthreads();

    if (kb <= Qr + 15) {   // wave-uniform causal tile skip: EXEC stays all-1s
      const _Float16* kv = lds_kv[kt & 1];
      // S = Q * K^T : 16 rows x 32 keys
      v8f s0 = {}, s1 = {};
      #pragma unroll
      for (int f = 0; f < 2; ++f) {
        const v16h bk0 = *(const v16h*)&kv[((f * 2 + 0) * 32 + lane) << 4];
        const v16h bk1 = *(const v16h*)&kv[((f * 2 + 1) * 32 + lane) << 4];
        s0 = wmma_f32_f16(f ? aq1 : aq0, bk0, s0);
        s1 = wmma_f32_f16(f ? aq1 : aq0, bk1, s1);
      }
      // causal mask + online softmax
      float corr[8];
      #pragma unroll
      for (int r = 0; r < 8; ++r) {
        int M = r + hi * 8;
        int qrow = Qr + M;
        float v0 = s0[r] * scale;
        float v1 = s1[r] * scale;
        if (kb + ln > qrow)      v0 = -1e30f;
        if (kb + 16 + ln > qrow) v1 = -1e30f;
        float rm = fmaxf(v0, v1);
        #pragma unroll
        for (int off = 1; off < 16; off <<= 1)
          rm = fmaxf(rm, __shfl_xor(rm, off, 32));
        float mnew = fmaxf(mrow[r], rm);
        corr[r] = __expf(mrow[r] - mnew);
        float p0 = __expf(v0 - mnew);
        float p1 = __expf(v1 - mnew);
        float rs = p0 + p1;
        #pragma unroll
        for (int off = 1; off < 16; off <<= 1)
          rs += __shfl_xor(rs, off, 32);
        lrow[r] = lrow[r] * corr[r] + rs;
        mrow[r] = mnew;
        // P into per-wave LDS directly in A-fragment order
        lds_pw[wid][(((ln >> 3) * 16 + M) << 4) + (ln & 7)]     = (_Float16)p0;
        lds_pw[wid][(((ln >> 3) * 16 + M) << 4) + (ln & 7) + 8] = (_Float16)p1;
      }
      #pragma unroll
      for (int n = 0; n < 4; ++n)
        #pragma unroll
        for (int r = 0; r < 8; ++r) accO[n][r] *= corr[r];

      // O += P * V
      const v16h ap = *(const v16h*)&lds_pw[wid][lane << 4];
      #pragma unroll
      for (int nt = 0; nt < 4; ++nt) {
        const v16h bv = *(const v16h*)&kv[2048 + ((nt * 32 + lane) << 4)];
        accO[nt] = wmma_f32_f16(ap, bv, accO[nt]);
      }
    }
    __syncthreads();
  }

  // epilogue: normalize rows and store f32
  #pragma unroll
  for (int r = 0; r < 8; ++r) {
    float inv = 1.0f / lrow[r];
    int grow = bb + Qr + r + hi * 8;
    #pragma unroll
    for (int nt = 0; nt < 4; ++nt)
      out[(size_t)grow * H_ + nt * 16 + ln] = accO[nt][r] * inv;
  }
}

extern "C" void kernel_launch(void* const* d_in, const int* in_sizes, int n_in,
                              void* d_out, int out_size, void* d_ws, size_t ws_size,
                              hipStream_t stream) {
  const float* x  = (const float*)d_in[0];
  const float* wq = (const float*)d_in[1];
  const float* wk = (const float*)d_in[2];
  const float* wv = (const float*)d_in[3];

  _Float16* ws   = (_Float16*)d_ws;
  _Float16* wsw  = ws;                       // 196608 halfs (swizzled weights)
  _Float16* qsw  = wsw + 196608;             // 1048576 halfs (Q, A-frag layout)
  _Float16* kvsw = qsw + 1048576;            // 2097152 halfs (K|V interleaved tiles)

  swz_w_kernel<<<768, 256, 0, stream>>>(wq, wk, wv, wsw);
  qkv_kernel<<<(B_ * S_) / 128, 256, 0, stream>>>(x, wsw, qsw, kvsw);
  attn_kernel<<<dim3(S_ / 128, B_), 256, 0, stream>>>(qsw, kvsw, (float*)d_out);
}
